// crowdFormerBlock_54425825575596
// MI455X (gfx1250) — compile-verified, matrix operand order checked by reading, not run on hardware
//
#include <hip/hip_runtime.h>
#include <hip/hip_bf16.h>
#include <math.h>

// ---------------------------------------------------------------------------
// CrowdFormer block for MI455X (gfx1250): bf16 WMMA everywhere, fp32 accum.
// L = 4096, D = 576, heads = 8, dh = 72 (padded: K->96, N->80), OUT_C = 128.
// ---------------------------------------------------------------------------

typedef __attribute__((ext_vector_type(16))) __bf16 v16bf;
typedef __attribute__((ext_vector_type(8)))  __bf16 v8bf;
typedef __attribute__((ext_vector_type(8)))  float  v8f;

#define L_TOK   4096
#define DMODEL  576
#define NHEAD   8
#define DH      72
#define KPAD    96     // dh padded to multiple of 32 for WMMA K
#define NPAD    80     // dh padded to multiple of 16 for O columns
#define OUTC    128
#define QK_SCALE 0.11785113019775793f  // 72^-0.5

// ---- WMMA helpers ----------------------------------------------------------

__device__ __forceinline__ v8f wmma_bf16(v16bf a, v16bf b, v8f c) {
  return __builtin_amdgcn_wmma_f32_16x16x32_bf16(
      false, a, false, b, (short)0, c, false, false);
}

// Load a 16x32 bf16 tile fragment (A layout; B^T uses the mirrored identical
// addressing) from a row-major matrix with leading dimension `ld` elements.
// Lane r=lane&15 owns row r; lane-group g=lane>>4 selects K chunks
// {g*8..g*8+7} and {16+g*8..16+g*8+7}; each chunk is one 16-byte load.
__device__ __forceinline__ v16bf load_frag(const __bf16* __restrict__ base,
                                           int ld, int lane) {
  const int r = lane & 15, g = lane >> 4;
  const __bf16* p = base + (size_t)r * ld + g * 8;
  v8bf lo = *(const v8bf*)(p);
  v8bf hi = *(const v8bf*)(p + 16);
  v16bf f;
#pragma unroll
  for (int i = 0; i < 8; ++i) { f[i] = lo[i]; f[8 + i] = hi[i]; }
  return f;
}

// ---- Stage 0: unfold 3x3 stride-2 pad-1 into x [4096 x 576] ---------------

__global__ void unfold_kernel(const float* __restrict__ fea,
                              float* __restrict__ xf,
                              __bf16* __restrict__ xb) {
  int idx = blockIdx.x * 256 + threadIdx.x;
  if (idx >= L_TOK * DMODEL) return;
  int l = idx / DMODEL, d = idx - l * DMODEL;
  int ho = l >> 6, wo = l & 63;
  int c = d / 9, k = d - c * 9;
  int ki = k / 3, kj = k - ki * 3;
  int hi = 2 * ho - 1 + ki, wi = 2 * wo - 1 + kj;
  float v = 0.f;
  if ((unsigned)hi < 128u && (unsigned)wi < 128u)
    v = fea[(c * 128 + hi) * 128 + wi];
  xf[idx] = v;
  xb[idx] = (__bf16)v;
}

// ---- Weight prep -----------------------------------------------------------

__global__ void transpose_to_bf16(const float* __restrict__ w,
                                  __bf16* __restrict__ wT, int R, int C) {
  int i = blockIdx.x * 256 + threadIdx.x;
  if (i >= R * C) return;
  int r = i / C, c = i - r * C;
  wT[(size_t)c * R + r] = (__bf16)w[i];
}

__global__ void convert_bf16(const float* __restrict__ w,
                             __bf16* __restrict__ o, int n) {
  int i = blockIdx.x * 256 + threadIdx.x;
  if (i < n) o[i] = (__bf16)w[i];
}

__global__ void zero_bf16(__bf16* __restrict__ p, int n) {
  int i = blockIdx.x * 256 + threadIdx.x;
  if (i < n) p[i] = (__bf16)0.f;
}

// ---- Generic WMMA GEMM: C[M,N] = A[M,K] @ Bt[N,K]^T, 3 epilogues ----------
// Workgroup: 256 threads = 8 waves (4 M x 2 N), each wave 32x32, WG 128x64.

template <int EPI>
__global__ void __launch_bounds__(256)
gemm_kernel(const __bf16* __restrict__ A, const __bf16* __restrict__ Bt,
            int M, int N, int K,
            const float* __restrict__ bias,   // [N] or null
            const float* __restrict__ resid,  // [M,N] fp32 (EPI==1)
            __bf16* __restrict__ outB,        // bf16 out (EPI==1)
            float* __restrict__ outF,         // final out (EPI==2)
            __bf16* __restrict__ qp, __bf16* __restrict__ kp,
            __bf16* __restrict__ vt) {
  const int lane = threadIdx.x & 31;
  const int wave = threadIdx.x >> 5;
  const int wm = wave & 3, wn = wave >> 2;
  const int m0 = blockIdx.x * 128 + wm * 32;
  const int n0 = blockIdx.y * 64 + wn * 32;

  v8f c00 = {}, c01 = {}, c10 = {}, c11 = {};
  for (int k0 = 0; k0 < K; k0 += 32) {
    v16bf a0 = load_frag(A + (size_t)m0 * K + k0, K, lane);
    v16bf a1 = load_frag(A + (size_t)(m0 + 16) * K + k0, K, lane);
    v16bf b0 = load_frag(Bt + (size_t)n0 * K + k0, K, lane);
    v16bf b1 = load_frag(Bt + (size_t)(n0 + 16) * K + k0, K, lane);
    c00 = wmma_bf16(a0, b0, c00);
    c01 = wmma_bf16(a0, b1, c01);
    c10 = wmma_bf16(a1, b0, c10);
    c11 = wmma_bf16(a1, b1, c11);
  }

  const int cl = lane & 15, g = lane >> 4;
#pragma unroll
  for (int t = 0; t < 4; ++t) {
    v8f c = (t == 0) ? c00 : (t == 1) ? c01 : (t == 2) ? c10 : c11;
    int rbase = m0 + (t >> 1) * 16 + 8 * g;
    int col = n0 + (t & 1) * 16 + cl;
#pragma unroll
    for (int r = 0; r < 8; ++r) {
      int row = rbase + r;
      float v = c[r];
      if (EPI == 0) {
        // QKV epilogue: bias, split q/k/v, pad, fold softmax scale into q,
        // store v transposed for contiguous B fragments in attn*V.
        v += bias[col];
        int which = col / DMODEL;
        int d = col - which * DMODEL;
        int h = d / DH, dd = d - h * DH;
        if (which == 0)
          qp[((size_t)h * L_TOK + row) * KPAD + dd] = (__bf16)(v * QK_SCALE);
        else if (which == 1)
          kp[((size_t)h * L_TOK + row) * KPAD + dd] = (__bf16)v;
        else
          vt[((size_t)h * NPAD + dd) * L_TOK + row] = (__bf16)v;
      } else if (EPI == 1) {
        // out-projection: + bias + fp32 residual -> bf16
        v += bias[col] + resid[(size_t)row * N + col];
        outB[(size_t)row * N + col] = (__bf16)v;
      } else {
        // final projection: SiLU, fold to [OUT_C][Ho*Wo] (column-major store)
        float s = v / (1.0f + __expf(-v));
        outF[(size_t)col * M + row] = s;
      }
    }
  }
}

// ---- Flash attention: wave = 16 query rows of one head --------------------
// Block 128 threads = 4 waves; grid (L/64, heads).

__global__ void __launch_bounds__(128)
attn_kernel(const __bf16* __restrict__ qp, const __bf16* __restrict__ kp,
            const __bf16* __restrict__ vt, __bf16* __restrict__ attn) {
  __shared__ __align__(16) __bf16 pstage[4][16 * 32];
  const int lane = threadIdx.x & 31;
  const int wave = threadIdx.x >> 5;
  const int h = blockIdx.y;
  const int q0 = blockIdx.x * 64 + wave * 16;
  const int cl = lane & 15, g = lane >> 4;

  const __bf16* qbase = qp + ((size_t)h * L_TOK + q0) * KPAD;
  v16bf aq0 = load_frag(qbase, KPAD, lane);
  v16bf aq1 = load_frag(qbase + 32, KPAD, lane);
  v16bf aq2 = load_frag(qbase + 64, KPAD, lane);

  float m[8], l[8];
  v8f o[5];
#pragma unroll
  for (int r = 0; r < 8; ++r) { m[r] = -1e30f; l[r] = 0.f; }
#pragma unroll
  for (int t = 0; t < 5; ++t) o[t] = v8f{};

  __bf16* pst = &pstage[wave][0];
  const __bf16* vhead = vt + (size_t)h * NPAD * L_TOK;

  for (int kb = 0; kb < L_TOK; kb += 32) {
    const __bf16* kb0 = kp + ((size_t)h * L_TOK + kb) * KPAD;
    // S = q @ k^T for two 16-key tiles, K = 96 (3 x 32)
    v8f s0 = {}, s1 = {};
    {
      v16bf b;
      b = load_frag(kb0, KPAD, lane);            s0 = wmma_bf16(aq0, b, s0);
      b = load_frag(kb0 + 32, KPAD, lane);       s0 = wmma_bf16(aq1, b, s0);
      b = load_frag(kb0 + 64, KPAD, lane);       s0 = wmma_bf16(aq2, b, s0);
      b = load_frag(kb0 + 16 * KPAD, KPAD, lane);      s1 = wmma_bf16(aq0, b, s1);
      b = load_frag(kb0 + 16 * KPAD + 32, KPAD, lane); s1 = wmma_bf16(aq1, b, s1);
      b = load_frag(kb0 + 16 * KPAD + 64, KPAD, lane); s1 = wmma_bf16(aq2, b, s1);
    }
    // Online softmax per row; row r lives across the 16 lanes of group g.
#pragma unroll
    for (int r = 0; r < 8; ++r) {
      float pm = fmaxf(s0[r], s1[r]);
      pm = fmaxf(pm, __shfl_xor(pm, 1));
      pm = fmaxf(pm, __shfl_xor(pm, 2));
      pm = fmaxf(pm, __shfl_xor(pm, 4));
      pm = fmaxf(pm, __shfl_xor(pm, 8));
      float mn = fmaxf(m[r], pm);
      float alpha = __expf(m[r] - mn);
      float p0 = __expf(s0[r] - mn);
      float p1 = __expf(s1[r] - mn);
      float rs = p0 + p1;
      rs += __shfl_xor(rs, 1);
      rs += __shfl_xor(rs, 2);
      rs += __shfl_xor(rs, 4);
      rs += __shfl_xor(rs, 8);
      l[r] = l[r] * alpha + rs;
      m[r] = mn;
#pragma unroll
      for (int t = 0; t < 5; ++t) o[t][r] *= alpha;
      // Stage P (16x32, row-major) in per-wave LDS to re-load in A layout.
      pst[(r + 8 * g) * 32 + cl]      = (__bf16)p0;
      pst[(r + 8 * g) * 32 + 16 + cl] = (__bf16)p1;
    }
    asm volatile("s_wait_dscnt 0" ::: "memory");
    v16bf pf = load_frag(pst, 32, lane);
    const __bf16* vb = vhead + kb;
#pragma unroll
    for (int t = 0; t < 5; ++t) {
      v16bf b = load_frag(vb + (size_t)t * 16 * L_TOK, L_TOK, lane);
      o[t] = wmma_bf16(pf, b, o[t]);
    }
  }

  // Normalize and write O back to attn [L, 576] (only real dh columns).
#pragma unroll
  for (int t = 0; t < 5; ++t)
#pragma unroll
    for (int r = 0; r < 8; ++r) {
      int col = t * 16 + cl;
      if (col < DH)
        attn[((size_t)(q0 + r + 8 * g)) * DMODEL + h * DH + col] =
            (__bf16)(o[t][r] / l[r]);
    }
}

// ---------------------------------------------------------------------------

extern "C" void kernel_launch(void* const* d_in, const int* in_sizes, int n_in,
                              void* d_out, int out_size, void* d_ws,
                              size_t ws_size, hipStream_t stream) {
  const float* fea    = (const float*)d_in[0];
  const float* w_qkv  = (const float*)d_in[1];
  const float* b_qkv  = (const float*)d_in[2];
  const float* w_out  = (const float*)d_in[3];
  const float* b_out  = (const float*)d_in[4];
  const float* conv_w = (const float*)d_in[5];
  float* out = (float*)d_out;

  // Workspace carve (~44 MB total)
  char* p = (char*)d_ws;
  auto carve = [&](size_t bytes) -> char* {
    char* q = p;
    p += (bytes + 255) & ~(size_t)255;
    return q;
  };
  float*  xf     = (float*)carve((size_t)L_TOK * DMODEL * 4);
  __bf16* xb     = (__bf16*)carve((size_t)L_TOK * DMODEL * 2);
  __bf16* wqkvT  = (__bf16*)carve((size_t)3 * DMODEL * DMODEL * 2);
  __bf16* woutT  = (__bf16*)carve((size_t)DMODEL * DMODEL * 2);
  __bf16* convw  = (__bf16*)carve((size_t)OUTC * DMODEL * 2);
  __bf16* qpad   = (__bf16*)carve((size_t)NHEAD * L_TOK * KPAD * 2);
  __bf16* kpad   = (__bf16*)carve((size_t)NHEAD * L_TOK * KPAD * 2);
  __bf16* vT     = (__bf16*)carve((size_t)NHEAD * NPAD * L_TOK * 2);
  __bf16* attnB  = (__bf16*)carve((size_t)L_TOK * DMODEL * 2);
  __bf16* resB   = (__bf16*)carve((size_t)L_TOK * DMODEL * 2);

  auto nb = [](int n) { return (n + 255) / 256; };

  // Stage 0: unfold + weight prep + zero padded buffers
  unfold_kernel<<<nb(L_TOK * DMODEL), 256, 0, stream>>>(fea, xf, xb);
  transpose_to_bf16<<<nb(DMODEL * 3 * DMODEL), 256, 0, stream>>>(
      w_qkv, wqkvT, DMODEL, 3 * DMODEL);
  transpose_to_bf16<<<nb(DMODEL * DMODEL), 256, 0, stream>>>(
      w_out, woutT, DMODEL, DMODEL);
  convert_bf16<<<nb(OUTC * DMODEL), 256, 0, stream>>>(conv_w, convw,
                                                      OUTC * DMODEL);
  zero_bf16<<<nb(NHEAD * L_TOK * KPAD), 256, 0, stream>>>(
      qpad, NHEAD * L_TOK * KPAD);
  zero_bf16<<<nb(NHEAD * L_TOK * KPAD), 256, 0, stream>>>(
      kpad, NHEAD * L_TOK * KPAD);
  zero_bf16<<<nb(NHEAD * NPAD * L_TOK), 256, 0, stream>>>(
      vT, NHEAD * NPAD * L_TOK);

  // Stage 1: QKV GEMM [4096 x 1728], scatter into padded q/k/vT
  {
    dim3 grid(L_TOK / 128, (3 * DMODEL) / 64);
    gemm_kernel<0><<<grid, 256, 0, stream>>>(xb, wqkvT, L_TOK, 3 * DMODEL,
                                             DMODEL, b_qkv, nullptr, nullptr,
                                             nullptr, qpad, kpad, vT);
  }

  // Stage 2: flash attention
  {
    dim3 grid(L_TOK / 64, NHEAD);
    attn_kernel<<<grid, 128, 0, stream>>>(qpad, kpad, vT, attnB);
  }

  // Stage 3: out projection + bias + residual -> bf16
  {
    dim3 grid(L_TOK / 128, DMODEL / 64);
    gemm_kernel<1><<<grid, 256, 0, stream>>>(attnB, woutT, L_TOK, DMODEL,
                                             DMODEL, b_out, xf, resB, nullptr,
                                             nullptr, nullptr, nullptr);
  }

  // Stage 4: final projection + SiLU + fold -> d_out [128][64][64]
  {
    dim3 grid(L_TOK / 128, OUTC / 64);
    gemm_kernel<2><<<grid, 256, 0, stream>>>(resB, convw, L_TOK, OUTC, DMODEL,
                                             nullptr, nullptr, nullptr, out,
                                             nullptr, nullptr, nullptr);
  }
}
